// FOLDSNet_75505525064284
// MI455X (gfx1250) — matrix operand order repaired and checked by reading, not compiled, not run on hardware
//
#include <hip/hip_runtime.h>
#include <hip/hip_bf16.h>
#include <math.h>

// ---- problem constants (mirror reference) ----
#define BATCH 256
#define CC 3
#define HH 224
#define WW 224
#define N_RET 128
#define N_LGN 128
#define N_V1  256
#define N_IT  128
#define N_CLASSES 1000
#define N_CLS_PAD 1024
#define INV_TEMP (1.0f/0.8f)

typedef __attribute__((ext_vector_type(16))) _Float16 v16h;
typedef __attribute__((ext_vector_type(8)))  _Float16 v8h;
typedef __attribute__((ext_vector_type(8)))  float    v8f;

__device__ __forceinline__ float sigmoidf(float v) {
    return 1.0f / (1.0f + expf(-v));
}

// ---------------------------------------------------------------------------
// Kernel 1: retina (pixel gather + MPJRD neuron) then LGN neuron -> r2 [B,128]
// ---------------------------------------------------------------------------
__global__ void k_ret_lgn(const float* __restrict__ x,      // [B,3,224,224]
                          const int*   __restrict__ pix,    // [128,16]
                          const float* __restrict__ rw,     // [128,4,4]
                          const float* __restrict__ rb,     // [128]
                          const float* __restrict__ lw,     // [128,4,4]
                          const float* __restrict__ lb,     // [128]
                          float* __restrict__ r2)           // [B,128]
{
    int idx = blockIdx.x * blockDim.x + threadIdx.x;        // b*128 + n
    if (idx >= BATCH * N_RET) return;
    int b = idx >> 7;
    int n = idx & 127;

    const float* xb = x + (size_t)b * (CC * HH * WW);
    const int*   pm = pix + n * 16;

    // Retina neuron: 4 dendrites x 4 synapses on gathered pixels
    float soma = 0.0f;
    #pragma unroll
    for (int d = 0; d < 4; ++d) {
        float acc = 0.0f;
        #pragma unroll
        for (int s = 0; s < 4; ++s)
            acc += xb[pm[d * 4 + s]] * rw[(n * 4 + d) * 4 + s];
        soma += tanhf(acc);
    }
    float r1 = sigmoidf(soma + rb[n]);

    // LGN neuron: input is r1 broadcast over 4x4 -> dend = tanh(r1 * sum_s w)
    float soma2 = 0.0f;
    #pragma unroll
    for (int d = 0; d < 4; ++d) {
        float wsum = 0.0f;
        #pragma unroll
        for (int s = 0; s < 4; ++s) wsum += lw[(n * 4 + d) * 4 + s];
        soma2 += tanhf(r1 * wsum);
    }
    r2[idx] = sigmoidf(soma2 + lb[n]);
}

// ---------------------------------------------------------------------------
// Kernel 2: V1 masked-softmax pool over lgn->v1 (Nin=128), then neuron (4x8)
// one block per batch element, 256 threads (one per V1 neuron)
// ---------------------------------------------------------------------------
__global__ void k_v1(const float* __restrict__ r2,          // [B,128]
                     const int*   __restrict__ mask,        // [256,128] 0/1
                     const float* __restrict__ vw,          // [256,4,8]
                     const float* __restrict__ vb,          // [256]
                     float* __restrict__ r3)                // [B,256]
{
    __shared__ float s[N_LGN];
    int b = blockIdx.x;
    int o = threadIdx.x;
    if (o < N_LGN) s[o] = r2[b * N_LGN + o];
    __syncthreads();

    // pack this output's mask row into 4 bitmask words
    const int* mrow = mask + o * N_LGN;
    unsigned mbits[4] = {0u, 0u, 0u, 0u};
    int cnt = 0;
    for (int n = 0; n < N_LGN; ++n) {
        if (mrow[n] != 0) { mbits[n >> 5] |= (1u << (n & 31)); ++cnt; }
    }
    if (cnt == 0) { mbits[0] = mbits[1] = mbits[2] = mbits[3] = 0xFFFFFFFFu; }

    float mx = -3.4e38f;
    for (int n = 0; n < N_LGN; ++n)
        if (mbits[n >> 5] & (1u << (n & 31))) mx = fmaxf(mx, s[n]);

    float z = 0.0f, num = 0.0f;
    for (int n = 0; n < N_LGN; ++n) {
        if (mbits[n >> 5] & (1u << (n & 31))) {
            float w = expf((s[n] - mx) * INV_TEMP);
            z += w; num += w * s[n];
        }
    }
    float agg = num / z;

    float soma = 0.0f;
    #pragma unroll
    for (int d = 0; d < 4; ++d) {
        float wsum = 0.0f;
        #pragma unroll
        for (int ss = 0; ss < 8; ++ss) wsum += vw[(o * 4 + d) * 8 + ss];
        soma += tanhf(agg * wsum);
    }
    r3[b * N_V1 + o] = sigmoidf(soma + vb[o]);
}

// ---------------------------------------------------------------------------
// Kernel 3: IT pool over v1->it (Nin=256), neuron (4x8), emit f16 r4 [B,128]
// one block per batch element, 128 threads
// ---------------------------------------------------------------------------
__global__ void k_it(const float* __restrict__ r3,          // [B,256]
                     const int*   __restrict__ mask,        // [128,256] 0/1
                     const float* __restrict__ iw,          // [128,4,8]
                     const float* __restrict__ ib,          // [128]
                     _Float16* __restrict__ r4h)            // [B,128] f16
{
    __shared__ float s[N_V1];
    int b = blockIdx.x;
    int o = threadIdx.x;
    s[o]       = r3[b * N_V1 + o];
    s[o + 128] = r3[b * N_V1 + o + 128];
    __syncthreads();

    const int* mrow = mask + o * N_V1;
    unsigned mbits[8] = {0u,0u,0u,0u,0u,0u,0u,0u};
    int cnt = 0;
    for (int n = 0; n < N_V1; ++n) {
        if (mrow[n] != 0) { mbits[n >> 5] |= (1u << (n & 31)); ++cnt; }
    }
    if (cnt == 0)
        for (int w = 0; w < 8; ++w) mbits[w] = 0xFFFFFFFFu;

    float mx = -3.4e38f;
    for (int n = 0; n < N_V1; ++n)
        if (mbits[n >> 5] & (1u << (n & 31))) mx = fmaxf(mx, s[n]);

    float z = 0.0f, num = 0.0f;
    for (int n = 0; n < N_V1; ++n) {
        if (mbits[n >> 5] & (1u << (n & 31))) {
            float w = expf((s[n] - mx) * INV_TEMP);
            z += w; num += w * s[n];
        }
    }
    float agg = num / z;

    float soma = 0.0f;
    #pragma unroll
    for (int d = 0; d < 4; ++d) {
        float wsum = 0.0f;
        #pragma unroll
        for (int ss = 0; ss < 8; ++ss) wsum += iw[(o * 4 + d) * 8 + ss];
        soma += tanhf(agg * wsum);
    }
    r4h[b * N_IT + o] = (_Float16)sigmoidf(soma + ib[o]);
}

// ---------------------------------------------------------------------------
// Kernel 4: cls_w f32 [1000,128] -> f16, zero padded to [1024,128]
// ---------------------------------------------------------------------------
__global__ void k_wcvt(const float* __restrict__ w, _Float16* __restrict__ wh)
{
    int i = blockIdx.x * blockDim.x + threadIdx.x;          // over 1024*128
    if (i >= N_CLS_PAD * N_IT) return;
    int c = i >> 7;
    wh[i] = (c < N_CLASSES) ? (_Float16)w[i] : (_Float16)0.0f;
}

// ---------------------------------------------------------------------------
// Kernel 5: classifier GEMM via v_wmma_f32_16x16x32_f16
// out[b,c] = sum_k r4[b,k] * cls_w[c,k] + cls_b[c]
// One wave32 per 16x16 output tile; K=128 -> 4 WMMAs per tile.
// CDNA5 layouts (ISA 7.12.2):
//   A 16x32 f16 : lanes 0-15 row M=l, halfs[0..7]=K0..7, [8..15]=K16..23
//                 lanes 16-31 row M=l-16, halfs[0..7]=K8..15, [8..15]=K24..31
//   B 32x16 f16 : lanes 0-15 col N=l, lanes 16-31 col N=l-16; K order same as A
//   C/D 16x16 f32: lanes 0-15 VGPRv=(M=v,N=l); lanes 16-31 VGPRv=(M=v+8,N=l-16)
// ---------------------------------------------------------------------------
__global__ void k_cls_gemm(const _Float16* __restrict__ Ah,   // [256,128]
                           const _Float16* __restrict__ Wh,   // [1024,128]
                           const float* __restrict__ bias,    // [1000]
                           float* __restrict__ out)           // [256,1000]
{
    int wave   = blockIdx.x * (blockDim.x >> 5) + (threadIdx.x >> 5);
    int lane   = threadIdx.x & 31;
    int tileM  = wave >> 6;      // 0..15  (batch / 16)
    int tileN  = wave & 63;      // 0..63  (padded classes / 16)
    int hi     = lane >> 4;      // lane half-group
    int l16    = lane & 15;

    const _Float16* arow = Ah + (size_t)(tileM * 16 + l16) * N_IT;
    const _Float16* brow = Wh + (size_t)(tileN * 16 + l16) * N_IT;

    v8f acc = {};
    #pragma unroll
    for (int kk = 0; kk < N_IT; kk += 32) {
        v8h a0 = *(const v8h*)(arow + kk + hi * 8);
        v8h a1 = *(const v8h*)(arow + kk + 16 + hi * 8);
        v16h A = __builtin_shufflevector(a0, a1,
                 0,1,2,3,4,5,6,7,8,9,10,11,12,13,14,15);
        v8h b0 = *(const v8h*)(brow + kk + hi * 8);
        v8h b1 = *(const v8h*)(brow + kk + 16 + hi * 8);
        v16h B = __builtin_shufflevector(b0, b1,
                 0,1,2,3,4,5,6,7,8,9,10,11,12,13,14,15);
        acc = __builtin_amdgcn_wmma_f32_16x16x32_f16(
                  false, A, false, B, (short)0, acc, false, false);
    }

    int ocol = tileN * 16 + l16;
    if (ocol < N_CLASSES) {
        float bb = bias[ocol];
        int rbase = tileM * 16 + hi * 8;
        #pragma unroll
        for (int v = 0; v < 8; ++v)
            out[(size_t)(rbase + v) * N_CLASSES + ocol] = acc[v] + bb;
    }
}

// ---------------------------------------------------------------------------
extern "C" void kernel_launch(void* const* d_in, const int* in_sizes, int n_in,
                              void* d_out, int out_size, void* d_ws, size_t ws_size,
                              hipStream_t stream)
{
    const float* x    = (const float*)d_in[0];
    const int*   pix  = (const int*)  d_in[1];
    const int*   m_lv = (const int*)  d_in[2];   // lgn_to_v1 [256,128]
    const int*   m_vi = (const int*)  d_in[3];   // v1_to_it  [128,256]
    const float* rw   = (const float*)d_in[4];
    const float* rb   = (const float*)d_in[5];
    const float* lw   = (const float*)d_in[6];
    const float* lb   = (const float*)d_in[7];
    const float* vw   = (const float*)d_in[8];
    const float* vb   = (const float*)d_in[9];
    const float* iw   = (const float*)d_in[10];
    const float* ib   = (const float*)d_in[11];
    const float* cw   = (const float*)d_in[12];
    const float* cb   = (const float*)d_in[13];
    float* out = (float*)d_out;

    // workspace layout (all offsets 256B aligned)
    char* ws = (char*)d_ws;
    float*    r2  = (float*)   (ws + 0);        // 256*128*4  = 131072
    float*    r3  = (float*)   (ws + 131072);   // 256*256*4  = 262144
    _Float16* r4h = (_Float16*)(ws + 393216);   // 256*128*2  =  65536
    _Float16* wh  = (_Float16*)(ws + 458752);   // 1024*128*2 = 262144  (end 720896)

    // 1) retina + lgn
    k_ret_lgn<<<(BATCH * N_RET + 255) / 256, 256, 0, stream>>>(
        x, pix, rw, rb, lw, lb, r2);

    // 2) V1 pool + neurons
    k_v1<<<BATCH, N_V1, 0, stream>>>(r2, m_lv, vw, vb, r3);

    // 3) IT pool + neurons (emits f16)
    k_it<<<BATCH, N_IT, 0, stream>>>(r3, m_vi, iw, ib, r4h);

    // 4) classifier weights -> f16 (padded)
    k_wcvt<<<(N_CLS_PAD * N_IT + 255) / 256, 256, 0, stream>>>(cw, wh);

    // 5) WMMA classifier GEMM: 16x64 tiles, 4 waves/block -> 256 blocks
    k_cls_gemm<<<(16 * 64) / 4, 128, 0, stream>>>(r4h, wh, cb, out);
}
